// ClassifierModel_60370060312681
// MI455X (gfx1250) — matrix operand order repaired and checked by reading, not compiled, not run on hardware
//
#include <hip/hip_runtime.h>
#include <hip/hip_bf16.h>

// GCN 2-layer forward on gfx1250 (wave32, WMMA).
// Pipeline: deg -> dinv -> cvt x->bf16 -> pack W1 (B-fragment order)
//           -> GEMM1 [v_wmma_f32_16x16x32_bf16, register-direct fragments,
//              fused self-loop init] -> edge scatter (f32 atomics, L2)
//           -> bias+relu+dropout -> GEMM2 (wave reduce, fused self-loop)
//           -> edge scatter -> softmax.
// Dims: N=100000, F=512, H=128, C=10, E=600000 (N,E derived from in_sizes).

typedef __attribute__((ext_vector_type(16))) __bf16 v16bf;
typedef __attribute__((ext_vector_type(8)))  float  v8f;

__device__ __forceinline__ unsigned f2bf_u(float f) {
  union { float f; unsigned u; } x; x.f = f;
  return (x.u + 0x7FFFu + ((x.u >> 16) & 1u)) >> 16;   // round-to-nearest-even
}

// ---------------- degree / normalization ----------------
__global__ void init_deg_k(float* deg, int N) {
  int i = blockIdx.x * blockDim.x + threadIdx.x;
  if (i < N) deg[i] = 1.0f;                            // self-loop
}

__global__ void count_deg_k(const long long* __restrict__ ei, float* deg,
                            long long E) {
  long long e = (long long)blockIdx.x * blockDim.x + threadIdx.x;
  if (e < E) atomicAdd(&deg[(int)ei[E + e]], 1.0f);
}

__global__ void dinv_k(float* deg, int N) {
  int i = blockIdx.x * blockDim.x + threadIdx.x;
  if (i < N) deg[i] = rsqrtf(deg[i]);                  // deg >= 1 always
}

// ---------------- precision prep ----------------
// x[N,512] f32 -> xbf[N,512] bf16, row-major (streaming, float4 in / uint2 out)
__global__ void cvt_x_k(const float* __restrict__ x,
                        unsigned short* __restrict__ xbf, int total4) {
  int t = blockIdx.x * blockDim.x + threadIdx.x;
  if (t >= total4) return;
  float4 v = ((const float4*)x)[t];
  uint2 p;
  p.x = f2bf_u(v.x) | (f2bf_u(v.y) << 16);
  p.y = f2bf_u(v.z) | (f2bf_u(v.w) << 16);
  ((uint2*)xbf)[t] = p;
}

// W1[512,128] f32 -> Bpack in WMMA B-fragment order (16-bit B 32x16 layout):
//   uint index i = ((s*8 + t)*32 + lane)*8 + j
//   holds bf16 pair (K, K+1) with K = 32*s + (lane<16?0:8) + 2*j + (j<4?0:8),
//   column n = 16*t + (lane&15).
__global__ void pack_W1_k(const float* __restrict__ W1,
                          unsigned* __restrict__ Bp) {
  int i = blockIdx.x * blockDim.x + threadIdx.x;
  if (i >= 16 * 8 * 32 * 8) return;
  int j    = i & 7;
  int lane = (i >> 3) & 31;
  int t    = (i >> 8) & 7;
  int s    = i >> 11;
  int n = t * 16 + (lane & 15);
  int k = s * 32 + ((lane < 16) ? 0 : 8) + 2 * j + ((j < 4) ? 0 : 8);
  Bp[i] = f2bf_u(W1[(size_t)k * 128 + n]) |
          (f2bf_u(W1[(size_t)(k + 1) * 128 + n]) << 16);
}

// ---------------- GEMM1: hm[N,128] = x @ W1 (bf16 WMMA, f32 acc) -----------
// Block = 256 threads = 8 waves; wave w owns the 16-col tile w of a 16-row
// block. A fragments: two b128 loads / K-step from row-major bf16 x
// (ISA 7.12.2: lanes 0-15 M=lane hold K 0..7 & 16..23, lanes 16-31 +8).
// B fragments: two b128 loads from prepacked W1 (L2-resident, coalesced).
// Fused epilogue also writes agg = hm * dinv^2 (self-loop term).
__global__ __launch_bounds__(256) void gemm1_wmma_k(
    const unsigned short* __restrict__ xbf, const unsigned* __restrict__ Bp,
    const float* __restrict__ dinv, float* __restrict__ hm,
    float* __restrict__ agg, int N) {
  const int lane = threadIdx.x & 31;
  const int wave = threadIdx.x >> 5;
  const int row0 = blockIdx.x * 16;
  const int m    = lane & 15;
  const int koff = (lane < 16) ? 0 : 8;

  int gr = row0 + m; if (gr > N - 1) gr = N - 1;        // clamp (stores guarded)
  const unsigned short* arow = xbf + (size_t)gr * 512 + koff;
  const unsigned*       bb   = Bp + ((size_t)wave * 32 + lane) * 8;

  v8f c = {0.f, 0.f, 0.f, 0.f, 0.f, 0.f, 0.f, 0.f};

  #pragma unroll
  for (int s = 0; s < 16; ++s) {
    union { v16bf v; uint4 q[2]; } a, b;
    a.q[0] = *(const uint4*)(arow + s * 32);            // K = 32s+koff .. +7
    a.q[1] = *(const uint4*)(arow + s * 32 + 16);       // K = 32s+16+koff .. +7
    b.q[0] = *(const uint4*)(bb + (size_t)s * 2048);
    b.q[1] = *(const uint4*)(bb + (size_t)s * 2048 + 4);
    if (s < 15)                                          // global_prefetch_b8
      __builtin_prefetch(arow + (s + 1) * 32, 0, 0);
    c = __builtin_amdgcn_wmma_f32_16x16x32_bf16(
        /*neg_a=*/false, a.v, /*neg_b=*/false, b.v,
        /*c_mod=*/(short)0, c, /*reuse_a=*/false, /*reuse_b=*/false);
  }

  // C layout: VGPR r -> (M = r + 8*(lane>=16), N = lane&15)
  const int n    = wave * 16 + (lane & 15);
  const int mofs = (lane < 16) ? 0 : 8;
  #pragma unroll
  for (int r = 0; r < 8; ++r) {
    int row = row0 + r + mofs;
    if (row < N) {
      float  val = c[r];
      float  sdv = dinv[row];
      size_t idx = (size_t)row * 128 + n;
      hm[idx]  = val;
      agg[idx] = val * sdv * sdv;
    }
  }
}

// ---------------- layer-1 edge scatter ----------------
__global__ __launch_bounds__(256) void scatter1_k(
    const long long* __restrict__ ei, const float* __restrict__ dinv,
    const float* __restrict__ hm, float* __restrict__ agg, long long E) {
  int lane = threadIdx.x & 31;
  long long e = (long long)blockIdx.x * 8 + (threadIdx.x >> 5); // wave per edge
  if (e >= E) return;
  int s = (int)ei[e], d = (int)ei[E + e];
  float nrm = dinv[s] * dinv[d];
  float4 v = ((const float4*)(hm + (size_t)s * 128))[lane];
  float* ap = agg + (size_t)d * 128 + lane * 4;
  atomicAdd(ap + 0, v.x * nrm);
  atomicAdd(ap + 1, v.y * nrm);
  atomicAdd(ap + 2, v.z * nrm);
  atomicAdd(ap + 3, v.w * nrm);
}

__global__ void bias_relu_drop_k(float* __restrict__ h,
                                 const float* __restrict__ b1,
                                 const float* __restrict__ drop_u, int N) {
  int t = blockIdx.x * blockDim.x + threadIdx.x;    // N*32 threads, float4 each
  if (t >= N * 32) return;
  int g = t & 31;
  float4 v  = ((float4*)h)[t];
  float4 bb = ((const float4*)b1)[g];
  float4 u  = ((const float4*)drop_u)[t];
  const float inv_keep = 1.0f / 0.9f;
  v.x = fmaxf(v.x + bb.x, 0.f) * ((u.x >= 0.1f) ? inv_keep : 0.f);
  v.y = fmaxf(v.y + bb.y, 0.f) * ((u.y >= 0.1f) ? inv_keep : 0.f);
  v.z = fmaxf(v.z + bb.z, 0.f) * ((u.z >= 0.1f) ? inv_keep : 0.f);
  v.w = fmaxf(v.w + bb.w, 0.f) * ((u.w >= 0.1f) ? inv_keep : 0.f);
  ((float4*)h)[t] = v;
}

// ------- GEMM2: hm2[N,10] = h @ W2[128,10], fused self-loop into out -------
__global__ __launch_bounds__(256) void gemm2_k(const float* __restrict__ h,
                                               const float* __restrict__ W2,
                                               const float* __restrict__ dinv,
                                               float* __restrict__ hm2,
                                               float* __restrict__ out, int N) {
  __shared__ float sW2[1280];
  for (int i = threadIdx.x; i < 1280; i += 256) sW2[i] = W2[i];
  __syncthreads();
  int lane = threadIdx.x & 31;
  int n = blockIdx.x * 8 + (threadIdx.x >> 5);      // wave per node
  if (n >= N) return;
  float4 hv = ((const float4*)(h + (size_t)n * 128))[lane];
  int k0 = lane * 4;
  float acc[10];
  #pragma unroll
  for (int cc = 0; cc < 10; ++cc) {
    acc[cc] = hv.x * sW2[(k0 + 0) * 10 + cc] + hv.y * sW2[(k0 + 1) * 10 + cc] +
              hv.z * sW2[(k0 + 2) * 10 + cc] + hv.w * sW2[(k0 + 3) * 10 + cc];
  }
  #pragma unroll
  for (int cc = 0; cc < 10; ++cc)
    #pragma unroll
    for (int off = 16; off > 0; off >>= 1)
      acc[cc] += __shfl_xor(acc[cc], off, 32);
  if (lane == 0) {
    float sdv = dinv[n]; sdv = sdv * sdv;
    #pragma unroll
    for (int cc = 0; cc < 10; ++cc) {
      hm2[(size_t)n * 10 + cc] = acc[cc];
      out[(size_t)n * 10 + cc] = acc[cc] * sdv;     // self-loop term
    }
  }
}

// ---------------- layer-2 edge scatter + softmax ----------------
__global__ void scatter2_k(const long long* __restrict__ ei,
                           const float* __restrict__ dinv,
                           const float* __restrict__ hm2,
                           float* __restrict__ out, long long E) {
  long long e = (long long)blockIdx.x * blockDim.x + threadIdx.x;
  if (e >= E) return;
  int s = (int)ei[e], d = (int)ei[E + e];
  float nrm = dinv[s] * dinv[d];
  const float* sp = hm2 + (size_t)s * 10;
  float* dp = out + (size_t)d * 10;
  #pragma unroll
  for (int cc = 0; cc < 10; ++cc) atomicAdd(dp + cc, sp[cc] * nrm);
}

__global__ void softmax_k(float* __restrict__ out,
                          const float* __restrict__ b2, int N) {
  int n = blockIdx.x * blockDim.x + threadIdx.x;
  if (n >= N) return;
  float v[10], mx = -INFINITY;
  #pragma unroll
  for (int cc = 0; cc < 10; ++cc) {
    v[cc] = out[(size_t)n * 10 + cc] + b2[cc];
    mx = fmaxf(mx, v[cc]);
  }
  float sum = 0.f;
  #pragma unroll
  for (int cc = 0; cc < 10; ++cc) { v[cc] = expf(v[cc] - mx); sum += v[cc]; }
  float inv = 1.0f / sum;
  #pragma unroll
  for (int cc = 0; cc < 10; ++cc) out[(size_t)n * 10 + cc] = v[cc] * inv;
}

// ---------------- host launcher ----------------
extern "C" void kernel_launch(void* const* d_in, const int* in_sizes, int n_in,
                              void* d_out, int out_size, void* d_ws, size_t ws_size,
                              hipStream_t stream) {
  const float*     x      = (const float*)d_in[0];
  const long long* ei     = (const long long*)d_in[1];
  const float*     drop_u = (const float*)d_in[2];
  const float*     W1     = (const float*)d_in[3];
  const float*     b1     = (const float*)d_in[4];
  const float*     W2     = (const float*)d_in[5];
  const float*     b2     = (const float*)d_in[6];
  float*           out    = (float*)d_out;

  const int       N = in_sizes[0] / 512;
  const long long E = in_sizes[1] / 2;

  // workspace layout (all 16B aligned):
  //   dinv[N] | xbf[N*512 bf16] | Bpack[32768 u32] | hm[N*128] | agg1[N*128] | hm2[N*10]
  char*           ws   = (char*)d_ws;
  float*          dinv = (float*)ws;
  unsigned short* xbf  = (unsigned short*)(ws + (size_t)N * 4);
  unsigned*       Bp   = (unsigned*)(ws + (size_t)N * 4 + (size_t)N * 1024);
  float*          hm   = (float*)((char*)Bp + 32768 * 4);
  float*          agg1 = hm + (size_t)N * 128;
  float*          hm2  = agg1 + (size_t)N * 128;

  const int T = 256;
  // 1) degrees with self-loops, then dinv = rsqrt(deg) in place
  init_deg_k <<<(N + T - 1) / T, T, 0, stream>>>(dinv, N);
  count_deg_k<<<(int)((E + T - 1) / T), T, 0, stream>>>(ei, dinv, E);
  dinv_k     <<<(N + T - 1) / T, T, 0, stream>>>(dinv, N);
  // 2) precision prep: x -> bf16 rows; W1 -> B-fragment pack (L2-resident)
  int total4 = N * 128;                         // N*512/4 float4 groups
  cvt_x_k    <<<(total4 + T - 1) / T, T, 0, stream>>>(x, xbf, total4);
  pack_W1_k  <<<(32768 + T - 1) / T, T, 0, stream>>>(W1, Bp);
  // 3) hm = x @ W1 (WMMA), agg1 = hm * dinv^2 fused
  gemm1_wmma_k<<<(N + 15) / 16, T, 0, stream>>>(xbf, Bp, dinv, hm, agg1, N);
  // 4) edge scatter into agg1
  scatter1_k <<<(int)((E + 7) / 8), T, 0, stream>>>(ei, dinv, hm, agg1, E);
  // 5) h = dropout(relu(agg1 + b1)) in place
  bias_relu_drop_k<<<(N * 32 + T - 1) / T, T, 0, stream>>>(agg1, b1, drop_u, N);
  // 6) hm2 = h @ W2; out = hm2 * dinv^2 fused
  gemm2_k<<<(N + 7) / 8, T, 0, stream>>>(agg1, W2, dinv, hm2, out, N);
  // 7) edge scatter into out, then softmax(out + b2)
  scatter2_k <<<(int)((E + T - 1) / T), T, 0, stream>>>(ei, dinv, hm2, out, E);
  softmax_k  <<<(N + T - 1) / T, T, 0, stream>>>(out, b2, N);
}
